// DecoderRNN_7361573945506
// MI455X (gfx1250) — matrix operand order repaired and checked by reading, not compiled, not run on hardware
//
#include <hip/hip_runtime.h>

#define DEV __device__ __forceinline__

typedef __attribute__((ext_vector_type(16))) __bf16 v16bf;
typedef __attribute__((ext_vector_type(8)))  float  v8f;
typedef __attribute__((ext_vector_type(4)))  int    v4i;

#define AS1 __attribute__((address_space(1)))
#define AS3 __attribute__((address_space(3)))

#if defined(__has_builtin)
#if __has_builtin(__builtin_amdgcn_global_load_async_to_lds_b128) && \
    __has_builtin(__builtin_amdgcn_s_wait_asynccnt)
#define USE_ASYNC_LDS 1
#endif
#endif
#ifndef USE_ASYNC_LDS
#define USE_ASYNC_LDS 0
#endif

union Frag {
  v16bf v;
  uint4 q[2];
};

// ---------------- helpers ----------------

DEV unsigned short f2bf(float f) {
  unsigned int u = __float_as_uint(f);
  u += 0x7FFFu + ((u >> 16) & 1u);   // round-to-nearest-even
  return (unsigned short)(u >> 16);
}

DEV float sigmoidf_(float x) { return 1.0f / (1.0f + __expf(-x)); }

// ---------------------------------------------------------------------------
// Block-cooperative GEMM core: 4-deep async-LDS ring, depth-2 DMA prefetch,
// software-pipelined A fragments, one barrier per K-chunk.
//  - Block = 256 threads = 8 waves; wave w owns output rows [w*16, w*16+16).
//  - Block output tile = 128 rows x (NS*16) cols.
//  - B panel ("N-major": Bt[n*ldb+k] == B[k][n]) staged in 32-K chunks via
//    global_load_async_to_lds_b128 into a 4-buffer LDS ring. Each wave issues
//    a uniform SI async instrs per stage; s_wait_asynccnt(2*SI) retires chunk
//    c while chunks c+1 and c+2 remain in flight (double-depth overlap).
//  - Ring-4 + single barrier/chunk is safe: stage(c+2) overwrites the buffer
//    last read by compute(c-2), and barrier(c-1) already separates them.
//  - Columns padded to 40-ushort stride (80B) -> conflict-free ds_load_b128.
//  - K is a template constant: full unroll, address math folds to immediates.
// LDS requirement: 4 * NS*16*40 ushorts.
// ---------------------------------------------------------------------------
template <int NS, int K>
DEV void gemm_block(const unsigned short* __restrict__ A,  int lda,   // block row 0
                    const unsigned short* __restrict__ Bt, int ldb,   // panel col 0
                    unsigned short* lds, v8f* acc) {
  constexpr int BUFSZ = NS * 16 * 40;      // ushorts per ring buffer
  constexpr int NU    = NS * 8;            // 16B staging units per wave per chunk
  constexpr int SI    = (NU + 31) / 32;    // async instrs per wave per stage
  constexpr int NC    = K / 32;            // K chunks
  static_assert(NC >= 2, "need at least two K chunks");

  const int lane = threadIdx.x & 31;
  const int wave = threadIdx.x >> 5;
  const int half = lane >> 4;   // 0/1
  const int l16  = lane & 15;
  // A lane layout (16-bit A 16x32): lanes 0-15 -> K={0..7,16..23}, 16-31 -> K={8..15,24..31}
  const unsigned short* ap = A + (size_t)(wave * 16 + l16) * lda + half * 8;

  // Per-thread staging descriptors, computed once.
  const unsigned short* gsrc[SI];
  int   loff[SI];
  bool  act[SI];
#pragma unroll
  for (int i = 0; i < SI; ++i) {
    int slot = i * 32 + lane;
    act[i] = slot < NU;
    int u = wave * NU + (act[i] ? slot : 0);
    int col = u >> 2, seg = u & 3;
    gsrc[i] = Bt + (size_t)col * ldb + seg * 8;
    loff[i] = col * 40 + seg * 8;
  }

  auto stage = [&](int c) {
    unsigned short* base = lds + (c & 3) * BUFSZ;
#pragma unroll
    for (int i = 0; i < SI; ++i) {
      if (act[i]) {
#if USE_ASYNC_LDS
        __builtin_amdgcn_global_load_async_to_lds_b128(
            (AS1 v4i*)(gsrc[i] + c * 32), (AS3 v4i*)(base + loff[i]), 0, 0);
#else
        *(uint4*)(base + loff[i]) = *(const uint4*)(gsrc[i] + c * 32);
#endif
      }
    }
  };

  auto loadA = [&](int c) {
    Frag a;
    a.q[0] = *(const uint4*)(ap + c * 32);
    a.q[1] = *(const uint4*)(ap + c * 32 + 16);
    __builtin_prefetch(ap + c * 32 + 64, 0, 1);       // global_prefetch_b8
    return a;
  };

  auto compute = [&](int c, const Frag& a) {
    const unsigned short* base = lds + (c & 3) * BUFSZ;
#pragma unroll
    for (int s = 0; s < NS; ++s) {
      Frag b;   // B lane layout (32x16): N = l16, K = half*8 + {0..7, 16..23}
      const unsigned short* bl = base + (s * 16 + l16) * 40 + half * 8;
      b.q[0] = *(const uint4*)(bl);
      b.q[1] = *(const uint4*)(bl + 16);
      acc[s] = __builtin_amdgcn_wmma_f32_16x16x32_bf16(
          false, a.v, false, b.v, (short)0, acc[s], false, false);
    }
  };

  // Prime: two chunks in flight, first A fragment in registers.
  stage(0);
  stage(1);
  Frag a_cur = loadA(0);

#pragma unroll
  for (int c = 0; c < NC - 2; ++c) {
    stage(c + 2);                                   // depth-2 DMA prefetch
#if USE_ASYNC_LDS
    __builtin_amdgcn_s_wait_asynccnt(2 * SI);       // chunk c landed
#endif
    __syncthreads();
    Frag a_nxt = loadA(c + 1);                      // hide A latency behind WMMAs
    compute(c, a_cur);
    a_cur = a_nxt;
  }
  // c = NC-2: no more stages; only chunk NC-1 may remain in flight.
#if USE_ASYNC_LDS
  __builtin_amdgcn_s_wait_asynccnt(SI);
#endif
  __syncthreads();
  {
    Frag a_nxt = loadA(NC - 1);
    compute(NC - 2, a_cur);
    a_cur = a_nxt;
  }
  // c = NC-1
#if USE_ASYNC_LDS
  __builtin_amdgcn_s_wait_asynccnt(0);
#endif
  __syncthreads();
  compute(NC - 1, a_cur);
  __syncthreads();                                  // caller may re-stage this LDS
}

// ---------------- problem constants ----------------
// EMBED=256, HIDDEN=512, VOCAB=10000, FEAT_DIM=1024, B=128, NREG=49, T=32

// Workspace layout (bytes)
constexpr size_t OFF_WFT   = 0;                         // W_ft bf16 (256,1024) N-major
constexpr size_t OFF_WCAT  = OFF_WFT   +   524288;      // WcatT bf16 (2048,512)
constexpr size_t OFF_UCAT  = OFF_WCAT  +  2097152;      // UcatT bf16 (2048,512)
constexpr size_t OFF_WFC   = OFF_UCAT  +  2097152;      // W_fc bf16 (10000,512)
constexpr size_t OFF_BCAT  = OFF_WFC   + 10240000;      // bcat f32 (2048)
constexpr size_t OFF_XMEAN = OFF_BCAT  +     8192;      // xmean bf16 (128,1024)
constexpr size_t OFF_FEBF  = OFF_XMEAN +   262144;      // feats_emb bf16 (128,256)
constexpr size_t OFF_XALL  = OFF_FEBF  +    65536;      // xall bf16 (32,128,512)
constexpr size_t OFF_HBF   = OFF_XALL  +  4194304;      // h bf16 (128,512)
constexpr size_t OFF_C     = OFF_HBF   +   131072;      // c f32 (128,512)
constexpr size_t OFF_GATES = OFF_C     +   262144;      // gates f32 (128,2048)
constexpr size_t OFF_HS    = OFF_GATES +  1048576;      // hs bf16 (32*128,512)

// ---------------- prep kernels ----------------

__global__ void k_conv_plain(const float* __restrict__ src,
                             unsigned short* __restrict__ dst, int n) {
  int i = blockIdx.x * blockDim.x + threadIdx.x;
  if (i < n) dst[i] = f2bf(src[i]);
}

// dst (2048,512) bf16: dst[n][k] = W_gate(n>>9)[k][n&511]   (each W is (512,512) row-major)
__global__ void k_pack_gatesT(const float* __restrict__ Wi, const float* __restrict__ Wf,
                              const float* __restrict__ Wc, const float* __restrict__ Wo,
                              unsigned short* __restrict__ dst) {
  int i = blockIdx.x * blockDim.x + threadIdx.x;
  if (i >= 2048 * 512) return;
  int n = i >> 9, k = i & 511;
  int gate = n >> 9, nl = n & 511;
  const float* W = (gate == 0) ? Wi : (gate == 1) ? Wf : (gate == 2) ? Wc : Wo;
  dst[i] = f2bf(W[k * 512 + nl]);
}

__global__ void k_bcat(const float* __restrict__ bi, const float* __restrict__ bf_,
                       const float* __restrict__ bc, const float* __restrict__ bo,
                       float* __restrict__ dst) {
  int n = blockIdx.x * blockDim.x + threadIdx.x;
  if (n >= 2048) return;
  int g = n >> 9, nl = n & 511;
  const float* b = (g == 0) ? bi : (g == 1) ? bf_ : (g == 2) ? bc : bo;
  dst[n] = b[nl];
}

// xmean[b][k] = mean_r features[b][r][k], stored bf16
__global__ void k_mean(const float* __restrict__ feat,
                       unsigned short* __restrict__ xmean) {
  int i = blockIdx.x * blockDim.x + threadIdx.x;
  if (i >= 128 * 1024) return;
  int b = i >> 10, k = i & 1023;
  const float* p = feat + (size_t)b * 49 * 1024 + k;
  float s = 0.f;
  for (int r = 0; r < 49; ++r) s += p[r * 1024];
  xmean[i] = f2bf(s * (1.0f / 49.0f));
}

// feats_emb = xmean(128,1024) @ W_ft.T(1024,256) + b_ft -> bf16 (128,256)
// grid = 16 col-tiles, block tile 128x16
__global__ void k_feats_emb(const unsigned short* __restrict__ xmean,
                            const unsigned short* __restrict__ wft,
                            const float* __restrict__ bft,
                            unsigned short* __restrict__ febf) {
  __shared__ unsigned short lds[4 * 16 * 40];
  v8f acc[1] = {};
  gemm_block<1, 1024>(xmean, 1024, wft + (size_t)blockIdx.x * 16 * 1024, 1024, lds, acc);
  int lane = threadIdx.x & 31, wave = threadIdx.x >> 5;
  int half = lane >> 4, l16 = lane & 15;
  int col = blockIdx.x * 16 + l16;
  float bias = bft[col];
  for (int r = 0; r < 8; ++r) {
    int m = wave * 16 + half * 8 + r;
    febf[m * 256 + col] = f2bf(acc[0][r] + bias);
  }
}

// xall[t][b][k] = k<256 ? bf16(emb[captions[b][t]][k]) : feats_emb_bf[b][k-256]
__global__ void k_xall(const int* __restrict__ captions, const float* __restrict__ emb,
                       const unsigned short* __restrict__ febf,
                       unsigned short* __restrict__ xall) {
  int i = blockIdx.x * blockDim.x + threadIdx.x;  // 32*128*512
  if (i >= 32 * 128 * 512) return;
  int k = i & 511;
  int b = (i >> 9) & 127;
  int t = i >> 16;
  unsigned short v;
  if (k < 256) {
    int tok = captions[b * 33 + t];
    v = f2bf(emb[(size_t)tok * 256 + k]);
  } else {
    v = febf[b * 256 + (k - 256)];
  }
  xall[i] = v;
}

__global__ void k_zero_state(unsigned short* __restrict__ hbf, float* __restrict__ c) {
  int i = blockIdx.x * blockDim.x + threadIdx.x;
  if (i >= 128 * 512) return;
  hbf[i] = 0;
  c[i] = 0.f;
}

// ---------------- LSTM step kernels ----------------

// gates(128,2048) = xt@WcatT' + h@UcatT' + bcat;  grid = 128 col-tiles, block 128x16
__global__ void k_gates(const unsigned short* __restrict__ xt,
                        const unsigned short* __restrict__ hbf,
                        const unsigned short* __restrict__ wcatT,
                        const unsigned short* __restrict__ ucatT,
                        const float* __restrict__ bcat,
                        float* __restrict__ gates) {
  __shared__ unsigned short lds[4 * 16 * 40];
  v8f acc[1] = {};
  gemm_block<1, 512>(xt,  512, wcatT + (size_t)blockIdx.x * 16 * 512, 512, lds, acc);
  gemm_block<1, 512>(hbf, 512, ucatT + (size_t)blockIdx.x * 16 * 512, 512, lds, acc);
  int lane = threadIdx.x & 31, wave = threadIdx.x >> 5;
  int half = lane >> 4, l16 = lane & 15;
  int col = blockIdx.x * 16 + l16;
  float bias = bcat[col];
  for (int r = 0; r < 8; ++r) {
    int m = wave * 16 + half * 8 + r;
    gates[m * 2048 + col] = acc[0][r] + bias;
  }
}

__global__ void k_cell(const float* __restrict__ gates, float* __restrict__ c,
                       unsigned short* __restrict__ hbf,
                       unsigned short* __restrict__ hs_t) {
  int i = blockIdx.x * blockDim.x + threadIdx.x;
  if (i >= 128 * 512) return;
  int b = i >> 9, n = i & 511;
  const float* g = gates + (size_t)b * 2048;
  float ig = sigmoidf_(g[n]);
  float fg = sigmoidf_(g[512 + n]);
  float gg = tanhf(g[1024 + n]);
  float og = sigmoidf_(g[1536 + n]);
  float cn = fg * c[i] + ig * gg;
  float hn = og * tanhf(cn);
  c[i] = cn;
  unsigned short hb = f2bf(hn);
  hbf[i] = hb;
  hs_t[i] = hb;
}

// ---------------- vocab projection ----------------
// out[b][t][v] = hs[t*128+b][:] . W_fc[v][:] + b_fc[v];  M=4096, K=512.
// Main: NS=8 -> block tile 128x128, grid (78, 32) covers cols [0, 9984).
// Tail: NS=1 -> block tile 128x16,  grid (1, 32) covers cols [9984, 10000).
template <int NS>
__global__ void k_vocab_t(const unsigned short* __restrict__ hs,
                          const unsigned short* __restrict__ wfc,
                          const float* __restrict__ bfc,
                          float* __restrict__ out, int col0) {
  __shared__ unsigned short lds[4 * NS * 16 * 40];
  v8f acc[NS] = {};
  int panel = col0 + blockIdx.x * NS * 16;
  const unsigned short* A  = hs  + (size_t)blockIdx.y * 128 * 512;
  const unsigned short* Bt = wfc + (size_t)panel * 512;
  gemm_block<NS, 512>(A, 512, Bt, 512, lds, acc);

  int lane = threadIdx.x & 31, wave = threadIdx.x >> 5;
  int half = lane >> 4, l16 = lane & 15;
#pragma unroll
  for (int s = 0; s < NS; ++s) {
    int col = panel + s * 16 + l16;
    float bias = bfc[col];
#pragma unroll
    for (int r = 0; r < 8; ++r) {
      int m = blockIdx.y * 128 + wave * 16 + half * 8 + r;
      int t = m >> 7, b = m & 127;
      __builtin_nontemporal_store(acc[s][r] + bias,
                                  out + (size_t)b * 320000 + (size_t)t * 10000 + col);
    }
  }
}

// ---------------- launch ----------------

extern "C" void kernel_launch(void* const* d_in, const int* in_sizes, int n_in,
                              void* d_out, int out_size, void* d_ws, size_t ws_size,
                              hipStream_t stream) {
  const float* features = (const float*)d_in[0];
  const int*   captions = (const int*)d_in[1];
  const float* emb   = (const float*)d_in[2];
  const float* W_ft  = (const float*)d_in[3];
  const float* b_ft  = (const float*)d_in[4];
  const float* W_i   = (const float*)d_in[5];
  const float* U_i   = (const float*)d_in[6];
  const float* b_i   = (const float*)d_in[7];
  const float* W_f   = (const float*)d_in[8];
  const float* U_f   = (const float*)d_in[9];
  const float* b_f   = (const float*)d_in[10];
  const float* W_c   = (const float*)d_in[11];
  const float* U_c   = (const float*)d_in[12];
  const float* b_c   = (const float*)d_in[13];
  const float* W_o   = (const float*)d_in[14];
  const float* U_o   = (const float*)d_in[15];
  const float* b_o   = (const float*)d_in[16];
  const float* W_fc  = (const float*)d_in[17];
  const float* b_fc  = (const float*)d_in[18];
  float* out = (float*)d_out;

  char* ws = (char*)d_ws;
  unsigned short* wft_bf = (unsigned short*)(ws + OFF_WFT);
  unsigned short* wcatT  = (unsigned short*)(ws + OFF_WCAT);
  unsigned short* ucatT  = (unsigned short*)(ws + OFF_UCAT);
  unsigned short* wfc_bf = (unsigned short*)(ws + OFF_WFC);
  float*          bcat   = (float*)(ws + OFF_BCAT);
  unsigned short* xmean  = (unsigned short*)(ws + OFF_XMEAN);
  unsigned short* febf   = (unsigned short*)(ws + OFF_FEBF);
  unsigned short* xall   = (unsigned short*)(ws + OFF_XALL);
  unsigned short* hbf    = (unsigned short*)(ws + OFF_HBF);
  float*          cst    = (float*)(ws + OFF_C);
  float*          gates  = (float*)(ws + OFF_GATES);
  unsigned short* hs     = (unsigned short*)(ws + OFF_HS);

  // --- weight conversion / packing (bf16) ---
  k_conv_plain<<<(256 * 1024 + 255) / 256, 256, 0, stream>>>(W_ft, wft_bf, 256 * 1024);
  k_conv_plain<<<(10000 * 512 + 255) / 256, 256, 0, stream>>>(W_fc, wfc_bf, 10000 * 512);
  k_pack_gatesT<<<(2048 * 512 + 255) / 256, 256, 0, stream>>>(W_i, W_f, W_c, W_o, wcatT);
  k_pack_gatesT<<<(2048 * 512 + 255) / 256, 256, 0, stream>>>(U_i, U_f, U_c, U_o, ucatT);
  k_bcat<<<8, 256, 0, stream>>>(b_i, b_f, b_c, b_o, bcat);

  // --- feature path ---
  k_mean<<<(128 * 1024 + 255) / 256, 256, 0, stream>>>(features, xmean);
  k_feats_emb<<<16, 256, 0, stream>>>(xmean, wft_bf, b_ft, febf);
  k_xall<<<(32 * 128 * 512 + 255) / 256, 256, 0, stream>>>(captions, emb, febf, xall);
  k_zero_state<<<(128 * 512 + 255) / 256, 256, 0, stream>>>(hbf, cst);

  // --- LSTM recurrence: 32 sequential steps ---
  for (int t = 0; t < 32; ++t) {
    k_gates<<<128, 256, 0, stream>>>(xall + (size_t)t * 128 * 512, hbf,
                                     wcatT, ucatT, bcat, gates);
    k_cell<<<(128 * 512 + 255) / 256, 256, 0, stream>>>(gates, cst, hbf,
                                                        hs + (size_t)t * 128 * 512);
  }

  // --- vocab projection: (4096,512) x (512,10000) ---
  dim3 vgrid(78, 32);
  k_vocab_t<8><<<vgrid, 256, 0, stream>>>(hs, wfc_bf, b_fc, out, 0);
  dim3 tgrid(1, 32);
  k_vocab_t<1><<<tgrid, 256, 0, stream>>>(hs, wfc_bf, b_fc, out, 9984);
}
